// View_selector_RI_34961033789527
// MI455X (gfx1250) — matrix operand order repaired and checked by reading, compiled, not run on hardware
//
#include <hip/hip_runtime.h>
#include <hip/hip_bf16.h>
#include <stdint.h>

typedef __attribute__((ext_vector_type(16))) __bf16 v16bf;
typedef __attribute__((ext_vector_type(8)))  float  v8f;

#define NVIEW 20
#define CDIM  512
#define HDIM  256
#define NCLS  40
#define SVIEW 8
#define KNN   4

__device__ __forceinline__ unsigned short f2bf(float f) {
    union { float f; unsigned int u; } x; x.f = f;
    unsigned int r = 0x7FFFu + ((x.u >> 16) & 1u);   // round-to-nearest-even
    return (unsigned short)((x.u + r) >> 16);
}

// ---------------------------------------------------------------------------
// Pack W1 (512x256) and W2 (256x40, padded to 48 cols) into bf16 WMMA
// B-fragment order: frag[ntile][kstep][lane][j], lane holds col = nt*16+(l&15),
// K_local = (j>=8?16:0)+(l>=16?8:0)+(j&7)  (ISA 7.12.2, 16-bit B 32x16 layout).
// ---------------------------------------------------------------------------
__global__ __launch_bounds__(256) void pack_weights(
    const float* __restrict__ W1, const float* __restrict__ W2,
    unsigned short* __restrict__ w1p, unsigned short* __restrict__ w2p)
{
    const int N1 = 16 * 16 * 32 * 16;   // 16 ntiles x 16 ksteps x 32 lanes x 16 halves
    const int N2 = 3 * 8 * 32 * 16;     // 3 ntiles x 8 ksteps
    int e = blockIdx.x * 256 + threadIdx.x;
    if (e < N1) {
        int j = e & 15, lane = (e >> 4) & 31, ks = (e >> 9) & 15, nt = e >> 13;
        int K   = ks * 32 + ((j >> 3) << 4) + ((lane >> 4) & 1) * 8 + (j & 7);
        int col = nt * 16 + (lane & 15);
        w1p[e] = f2bf(W1[K * HDIM + col]);
    } else {
        int e2 = e - N1;
        if (e2 < N2) {
            int j = e2 & 15, lane = (e2 >> 4) & 31, ks = (e2 >> 9) & 7, nt = e2 >> 12;
            int K   = ks * 32 + ((j >> 3) << 4) + ((lane >> 4) & 1) * 8 + (j & 7);
            int col = nt * 16 + (lane & 15);
            float v = (col < NCLS) ? W2[K * NCLS + col] : 0.0f;
            w2p[e2] = f2bf(v);
        }
    }
}

// ---------------------------------------------------------------------------
// Fused MLP: rows = B*20 of F0.  16 rows per block, 8 waves.
// GEMM1: (16x512)@(512x256) bf16 WMMA, +b1, leakyReLU -> H in LDS (bf16 frags)
// GEMM2: (16x256)@(256x48)  -> score_init (cols<40) + per-row max(softmax).
// A-fragment LDS layout (ISA 7.12.2, 16-bit A 16x32):
//   lane = row + 16*k[3], element j = (k&7) + 8*k[4], contiguous 32B per lane.
// ---------------------------------------------------------------------------
__global__ __launch_bounds__(256) void cls_gemm(
    const float* __restrict__ F0,
    const unsigned short* __restrict__ w1p,
    const unsigned short* __restrict__ w2p,
    const float* __restrict__ b1, const float* __restrict__ b2,
    float* __restrict__ out_scores,      // [rows][40]
    float* __restrict__ scoreRow)        // [rows] = max(softmax(row))
{
    __shared__ __align__(32) unsigned short ALDS[16 * 32 * 16]; // [ks][lane][16]
    __shared__ __align__(32) unsigned short HLDS[8 * 32 * 16];  // [ks][lane][16]
    __shared__ float SLDS[16 * NCLS];

    const int tid  = threadIdx.x;
    const int lane = tid & 31;
    const int wave = tid >> 5;
    const int blk  = blockIdx.x;
    const int row0 = blk * 16;

    // Cooperative load of A tile (16 rows x 512 f32) -> bf16 fragments in LDS
    for (int i = 0; i < 32; ++i) {
        int e   = i * 256 + tid;       // 0..8191, coalesced over tid
        int row = e >> 9;
        int k   = e & 511;
        float v = F0[(size_t)(row0 + row) * CDIM + k];
        int ks = k >> 5, kl = k & 31;
        int l2 = row + (((kl >> 3) & 1) << 4);
        int j  = (kl & 7) + (((kl >> 4) & 1) << 3);
        ALDS[(ks * 32 + l2) * 16 + j] = f2bf(v);
    }
    __syncthreads();

    // GEMM1: each wave owns n-tiles 2w and 2w+1 of the 256-wide H
    const int nt0 = wave * 2, nt1 = nt0 + 1;
    v8f acc0 = {0, 0, 0, 0, 0, 0, 0, 0};
    v8f acc1 = {0, 0, 0, 0, 0, 0, 0, 0};
    for (int ks = 0; ks < 16; ++ks) {
        v16bf a  = *(const v16bf*)&ALDS[(ks * 32 + lane) * 16];
        v16bf bA = *(const v16bf*)&w1p[((nt0 * 16 + ks) * 32 + lane) * 16];
        v16bf bB = *(const v16bf*)&w1p[((nt1 * 16 + ks) * 32 + lane) * 16];
        acc0 = __builtin_amdgcn_wmma_f32_16x16x32_bf16(false, a, false, bA,
                                                       (short)0, acc0, false, false);
        acc1 = __builtin_amdgcn_wmma_f32_16x16x32_bf16(false, a, false, bB,
                                                       (short)0, acc1, false, false);
    }

    // bias + leakyReLU + restage H as bf16 A-fragments for GEMM2
    const int hi = (lane >> 4) & 1;
    {
        int n0 = nt0 * 16 + (lane & 15);
        int n1 = nt1 * 16 + (lane & 15);
        float bb0 = b1[n0], bb1 = b1[n1];
        for (int i = 0; i < 8; ++i) {
            int m = i + hi * 8;
            float h0 = acc0[i] + bb0; h0 = (h0 >= 0.0f) ? h0 : 0.2f * h0;
            float h1 = acc1[i] + bb1; h1 = (h1 >= 0.0f) ? h1 : 0.2f * h1;
            int kl0 = n0 & 31, ks0 = n0 >> 5;
            HLDS[(ks0 * 32 + (m + (((kl0 >> 3) & 1) << 4))) * 16 +
                 ((kl0 & 7) + (((kl0 >> 4) & 1) << 3))] = f2bf(h0);
            int kl1 = n1 & 31, ks1 = n1 >> 5;
            HLDS[(ks1 * 32 + (m + (((kl1 >> 3) & 1) << 4))) * 16 +
                 ((kl1 & 7) + (((kl1 >> 4) & 1) << 3))] = f2bf(h1);
        }
    }
    __syncthreads();

    // GEMM2: waves 0..2 each compute one 16x16 tile (N padded 40 -> 48)
    if (wave < 3) {
        v8f acc2 = {0, 0, 0, 0, 0, 0, 0, 0};
        for (int ks = 0; ks < 8; ++ks) {
            v16bf a = *(const v16bf*)&HLDS[(ks * 32 + lane) * 16];
            v16bf b = *(const v16bf*)&w2p[((wave * 8 + ks) * 32 + lane) * 16];
            acc2 = __builtin_amdgcn_wmma_f32_16x16x32_bf16(false, a, false, b,
                                                           (short)0, acc2, false, false);
        }
        int n = wave * 16 + (lane & 15);
        bool valid = (n < NCLS);
        float bb = valid ? b2[n] : 0.0f;
        for (int i = 0; i < 8; ++i) {
            int m = i + hi * 8;
            float v = acc2[i] + bb;
            if (valid) {
                out_scores[(size_t)(row0 + m) * NCLS + n] = v;
                SLDS[m * NCLS + n] = v;
            }
        }
    }
    __syncthreads();

    // per-row score = max(softmax) = 1 / sum(exp(x - max))
    if (wave == 0 && lane < 16) {
        float mx = -1e30f;
        for (int j = 0; j < NCLS; ++j) mx = fmaxf(mx, SLDS[lane * NCLS + j]);
        float s = 0.0f;
        for (int j = 0; j < NCLS; ++j) s += __expf(SLDS[lane * NCLS + j] - mx);
        scoreRow[row0 + lane] = 1.0f / s;
    }
}

// ---------------------------------------------------------------------------
// Per-batch: argmax over views, view-0 swap perm, FPS (8 centers), kNN (k=4).
// One wave per batch element; lane l < 20 owns view l.  All argmax/argmin
// reductions use first-index tie-break to match jnp.argmax / lax.top_k.
// Output: rowsrc[b][s][kk] = b*20 + perm[id_knn]  (row index into F0/score).
// ---------------------------------------------------------------------------
__global__ __launch_bounds__(32) void select_fps_knn(
    const float* __restrict__ scoreRow,
    const float* __restrict__ vertices0,
    int* __restrict__ rowsrc)
{
    const int b = blockIdx.x;
    const int l = threadIdx.x;
    const bool act = (l < NVIEW);

    // id0 = argmax_v score
    float v  = act ? scoreRow[b * NVIEW + l] : -1e30f;
    int  idx = act ? l : 1000;
    for (int off = 16; off > 0; off >>= 1) {
        float ov = __shfl_xor(v, off);
        int   oi = __shfl_xor(idx, off);
        if (ov > v || (ov == v && oi < idx)) { v = ov; idx = oi; }
    }
    const int id0 = idx;

    // permuted vertex for this lane
    const int pl = (l == 0) ? id0 : ((l == id0) ? 0 : l);
    float px = 0.0f, py = 0.0f, pz = 0.0f;
    if (act) {
        const float* vp = vertices0 + (size_t)(b * NVIEW + pl) * 3;
        px = vp[0]; py = vp[1]; pz = vp[2];
    }

    // farthest point sampling (records incoming farthest, cents[0] = 0)
    float dist = act ? 1e10f : -1e30f;
    int far = 0;
    int cents[SVIEW];
#pragma unroll
    for (int it = 0; it < SVIEW; ++it) {
        cents[it] = far;
        float cx = __shfl(px, far), cy = __shfl(py, far), cz = __shfl(pz, far);
        if (act) {
            float dx = px - cx, dy = py - cy, dz = pz - cz;
            dist = fminf(dist, dx * dx + dy * dy + dz * dz);
        }
        float mv = dist; int mi = act ? l : 1000;
        for (int off = 16; off > 0; off >>= 1) {
            float ov = __shfl_xor(mv, off);
            int   oi = __shfl_xor(mi, off);
            if (ov > mv || (ov == mv && oi < mi)) { mv = ov; mi = oi; }
        }
        far = mi;
    }

    // kNN: 4 nearest (ascending distance, index tie-break) per sampled center
#pragma unroll
    for (int s = 0; s < SVIEW; ++s) {
        int c = cents[s];
        float cx = __shfl(px, c), cy = __shfl(py, c), cz = __shfl(pz, c);
        float dcur = 1e30f;
        if (act) {
            float dx = px - cx, dy = py - cy, dz = pz - cz;
            dcur = dx * dx + dy * dy + dz * dz;
        }
#pragma unroll
        for (int kk = 0; kk < KNN; ++kk) {
            float mv = dcur; int mi = act ? l : 1000;
            for (int off = 16; off > 0; off >>= 1) {
                float ov = __shfl_xor(mv, off);
                int   oi = __shfl_xor(mi, off);
                if (ov < mv || (ov == mv && oi < mi)) { mv = ov; mi = oi; }
            }
            int g = (mi == 0) ? id0 : ((mi == id0) ? 0 : mi);  // back to F0 view
            if (l == 0) rowsrc[(b * SVIEW + s) * KNN + kk] = b * NVIEW + g;
            if (l == mi) dcur = 1e30f;  // remove winner
        }
    }
}

// ---------------------------------------------------------------------------
// KEY FUSION: _cls is row-wise, so F_score[b,s,kk,:] == score_init gathered at
// rowsrc -- no second GEMM pass needed.  Softmax over the s-axis per (kk,j),
// F1_id = argmax_kk max_j softmax, then emit vertices_new + selected row ids.
// ---------------------------------------------------------------------------
__global__ __launch_bounds__(32) void choose_and_vertices(
    const float* __restrict__ score_init,   // [B*20][40], already written
    const int* __restrict__ rowsrc,         // [B][8][4]
    const float* __restrict__ vertices0,
    float* __restrict__ vout,               // [B][8][3]
    int* __restrict__ selrow)               // [B][8] -> row into F0
{
    __shared__ int   rows[SVIEW * KNN];
    __shared__ float mbuf[KNN * NCLS];
    __shared__ float sbuf[KNN * NCLS];

    const int b = blockIdx.x;
    const int l = threadIdx.x;

    if (l < SVIEW * KNN) rows[l] = rowsrc[b * SVIEW * KNN + l];
    __syncthreads();

    // per (kk,j): max and sum-exp over the 8 samples (softmax axis = s)
    for (int p = l; p < KNN * NCLS; p += 32) {
        int kk = p / NCLS, j = p % NCLS;
        float mx = -1e30f;
        for (int s = 0; s < SVIEW; ++s)
            mx = fmaxf(mx, score_init[(size_t)rows[s * KNN + kk] * NCLS + j]);
        float sm = 0.0f;
        for (int s = 0; s < SVIEW; ++s)
            sm += __expf(score_init[(size_t)rows[s * KNN + kk] * NCLS + j] - mx);
        mbuf[p] = mx; sbuf[p] = sm;
    }
    __syncthreads();

    if (l < SVIEW) {
        const int s = l;
        float best = -1e30f; int bestkk = 0;
        for (int kk = 0; kk < KNN; ++kk) {
            const float* r = score_init + (size_t)rows[s * KNN + kk] * NCLS;
            float f1 = -1e30f;
            for (int j = 0; j < NCLS; ++j) {
                float val = __expf(r[j] - mbuf[kk * NCLS + j]) / sbuf[kk * NCLS + j];
                f1 = fmaxf(f1, val);
            }
            if (f1 > best) { best = f1; bestkk = kk; }   // first-max tie-break
        }
        int sel = rows[s * KNN + bestkk];
        selrow[b * SVIEW + s] = sel;
        const float* vp = vertices0 + (size_t)sel * 3;
        float* o = vout + (size_t)(b * SVIEW + s) * 3;
        o[0] = vp[0]; o[1] = vp[1]; o[2] = vp[2];
    }
}

// F_new[b,s,:] = F0[selrow[b,s], :]   (512 floats / row, float4 coalesced)
__global__ __launch_bounds__(128) void gather_F(
    const float* __restrict__ F0, const int* __restrict__ selrow,
    float* __restrict__ Fnew)
{
    int row = blockIdx.x;
    int src = selrow[row];
    const float4* s = (const float4*)(F0 + (size_t)src * CDIM);
    float4*       d = (float4*)(Fnew + (size_t)row * CDIM);
    d[threadIdx.x] = s[threadIdx.x];
}

extern "C" void kernel_launch(void* const* d_in, const int* in_sizes, int n_in,
                              void* d_out, int out_size, void* d_ws, size_t ws_size,
                              hipStream_t stream)
{
    const float* F0        = (const float*)d_in[0];
    const float* vertices0 = (const float*)d_in[1];
    const float* W1        = (const float*)d_in[2];
    const float* b1        = (const float*)d_in[3];
    const float* W2        = (const float*)d_in[4];
    const float* b2        = (const float*)d_in[5];
    // d_in[6] is k (==4), baked in as KNN

    const int Bn = in_sizes[0] / (NVIEW * CDIM);   // 8192

    char* ws = (char*)d_ws;
    size_t off = 0;
    auto wsalloc = [&](size_t bytes) -> char* {
        char* p = ws + off;
        off = (off + bytes + 255) & ~(size_t)255;
        return p;
    };
    unsigned short* w1p     = (unsigned short*)wsalloc((size_t)16 * 16 * 32 * 16 * 2);
    unsigned short* w2p     = (unsigned short*)wsalloc((size_t)3 * 8 * 32 * 16 * 2);
    float*          scoreRw = (float*)wsalloc((size_t)Bn * NVIEW * sizeof(float));
    int*            rowsrc  = (int*)wsalloc((size_t)Bn * SVIEW * KNN * sizeof(int));
    int*            selrow  = (int*)wsalloc((size_t)Bn * SVIEW * sizeof(int));
    (void)ws_size; (void)n_in; (void)out_size;

    float* out_Fnew  = (float*)d_out;                                  // [B,8,512]
    float* out_score = out_Fnew + (size_t)Bn * SVIEW * CDIM;           // [B,20,40]
    float* out_vert  = out_score + (size_t)Bn * NVIEW * NCLS;          // [B,8,3]

    const int packN = 16 * 16 * 32 * 16 + 3 * 8 * 32 * 16;
    pack_weights<<<(packN + 255) / 256, 256, 0, stream>>>(W1, W2, w1p, w2p);

    cls_gemm<<<(Bn * NVIEW) / 16, 256, 0, stream>>>(
        F0, w1p, w2p, b1, b2, out_score, scoreRw);

    select_fps_knn<<<Bn, 32, 0, stream>>>(scoreRw, vertices0, rowsrc);

    choose_and_vertices<<<Bn, 32, 0, stream>>>(
        out_score, rowsrc, vertices0, out_vert, selrow);

    gather_F<<<Bn * SVIEW, 128, 0, stream>>>(F0, selrow, out_Fnew);
}
